// Model_72713796321670
// MI455X (gfx1250) — matrix operand order repaired and checked by reading, not compiled, not run on hardware
//
#include <hip/hip_runtime.h>
#include <hip/hip_bf16.h>
#include <stdint.h>

// Problem constants (from setup_inputs)
#define Bn    8
#define Cn    80
#define Hn    256
#define Wn    256
#define HWn   65536
#define Kn    100     // topK
#define KPAD  128     // padded to WMMA K multiple
#define NBIN  1024    // mantissa histogram bins
#define CAP   4096    // candidate cap per batch
#define TH    0.5f

typedef __attribute__((ext_vector_type(16))) _Float16 v16h;
typedef __attribute__((ext_vector_type(8)))  float    v8f;

// ---------------------------------------------------------------------------
// 3x3 max-pool NMS value at pixel (h,w) of one [256,256] plane.
// SAME padding with -inf => out-of-bounds neighbors ignored.
// keep iff center >= every in-bounds neighbor (matches hmax==hm).
// ---------------------------------------------------------------------------
__device__ __forceinline__ float nms_val(const float* __restrict__ plane, int p) {
    int h = p >> 8, w = p & 255;
    float v = plane[p];
    bool keep = true;
#pragma unroll
    for (int dy = -1; dy <= 1; ++dy) {
#pragma unroll
        for (int dx = -1; dx <= 1; ++dx) {
            if (dy == 0 && dx == 0) continue;
            int hh = h + dy, ww = w + dx;
            if (hh >= 0 && hh < Hn && ww >= 0 && ww < Wn) {
                keep = keep && (v >= plane[(hh << 8) | ww]);
            }
        }
    }
    return keep ? v : 0.0f;
}

// Monotonic bin from float bits; candidates are in (0.5, 1.0) -> exp 0x7E,
// so bin on mantissa. Clamped so any value >= 1.0 lands in the top bin.
__device__ __forceinline__ uint32_t bin_of(float v) {
    uint32_t bits = __float_as_uint(v);
    uint32_t b = (bits >= 0x3F000000u) ? ((bits - 0x3F000000u) >> 13) : 0u;
    return b > (NBIN - 1u) ? (NBIN - 1u) : b;
}

// ---------------------------------------------------------------------------
// Kernel 0: zero histogram + counters (ws is not re-poisoned between replays)
// ---------------------------------------------------------------------------
__global__ __launch_bounds__(1024) void k_init(uint32_t* __restrict__ hist) {
    // hist[Bn*NBIN] then thr[Bn] then cnt[Bn] are contiguous
    const int total = Bn * NBIN + 2 * Bn;
    for (int i = threadIdx.x; i < total; i += 1024) hist[i] = 0u;
}

// ---------------------------------------------------------------------------
// Kernel 1: NMS + per-batch histogram of values > TH. One block per (b,c)
// plane; LDS-local histogram, flushed with global atomics.
// ---------------------------------------------------------------------------
__global__ __launch_bounds__(1024) void k_nms_hist(const float* __restrict__ hm,
                                                   uint32_t* __restrict__ hist) {
    __shared__ uint32_t lh[NBIN];
    for (int i = threadIdx.x; i < NBIN; i += 1024) lh[i] = 0u;
    __syncthreads();

    int bc = blockIdx.x;
    int b  = bc / Cn;
    const float* plane = hm + (size_t)bc * HWn;

#pragma unroll 4
    for (int i = 0; i < HWn / 1024; ++i) {
        int p = i * 1024 + threadIdx.x;          // coalesced
        float v = nms_val(plane, p);
        if (v > TH) atomicAdd(&lh[bin_of(v)], 1u);
    }
    __syncthreads();
    for (int i = threadIdx.x; i < NBIN; i += 1024) {
        uint32_t c = lh[i];
        if (c) atomicAdd(&hist[b * NBIN + i], c);
    }
}

// ---------------------------------------------------------------------------
// Kernel 2: per-batch threshold bin T: largest bin with cum-count(from top)
// >= Kn. Anything in a bin < T has >= Kn strictly greater values above it.
// ---------------------------------------------------------------------------
__global__ void k_thresh(const uint32_t* __restrict__ hist,
                         uint32_t* __restrict__ thr) {
    int b = threadIdx.x;
    if (b < Bn) {
        uint32_t acc = 0, t = 0;
        for (int i = NBIN - 1; i >= 0; --i) {
            acc += hist[b * NBIN + i];
            if (acc >= Kn) { t = (uint32_t)i; break; }
        }
        thr[b] = t;
    }
}

// ---------------------------------------------------------------------------
// Kernel 3: collect candidates (bin >= T). hm is L2-resident after kernel 1
// (168MB < 192MB L2), so this pass streams from L2.
// ---------------------------------------------------------------------------
__global__ __launch_bounds__(1024) void k_collect(const float* __restrict__ hm,
                                                  const uint32_t* __restrict__ thr,
                                                  uint32_t* __restrict__ cnt,
                                                  float* __restrict__ cval,
                                                  uint32_t* __restrict__ cidx) {
    int bc = blockIdx.x;
    int b  = bc / Cn;
    int c  = bc % Cn;
    uint32_t T = thr[b];
    const float* plane = hm + (size_t)bc * HWn;

#pragma unroll 4
    for (int i = 0; i < HWn / 1024; ++i) {
        int p = i * 1024 + threadIdx.x;
        float v = nms_val(plane, p);
        if (v > TH && bin_of(v) >= T) {
            uint32_t pos = atomicAdd(&cnt[b], 1u);
            if (pos < CAP) {
                cval[b * CAP + pos] = v;
                cidx[b * CAP + pos] = (uint32_t)(c * HWn + p);  // flat (c,h,w)
            }
        }
    }
}

// ---------------------------------------------------------------------------
// Kernel 4: exact per-batch top-100 by (value desc, flat idx asc) via bitonic
// sort of 64-bit keys in LDS, then build padded box arrays. Invalid / padding
// slots become empty boxes (x1>x2, y1>y2) so rasterization needs no flags.
// ---------------------------------------------------------------------------
__global__ __launch_bounds__(256) void k_select(const float* __restrict__ cval,
                                                const uint32_t* __restrict__ cidx,
                                                const uint32_t* __restrict__ cnt,
                                                const float* __restrict__ off,
                                                const float* __restrict__ wh,
                                                float* __restrict__ boxes) {
    __shared__ unsigned long long key[CAP];
    int b = blockIdx.x;
    int n = (int)cnt[b];
    if (n > CAP) n = CAP;

    for (int i = threadIdx.x; i < CAP; i += 256) {
        unsigned long long k = 0ull;
        if (i < n) {
            k = ((unsigned long long)__float_as_uint(cval[b * CAP + i]) << 32)
              | (unsigned long long)(0xFFFFFFFFu - cidx[b * CAP + i]);
        }
        key[i] = k;
    }
    __syncthreads();

    // bitonic sort, final order descending (larger key = higher score /
    // smaller index on ties)
    for (int kk = 2; kk <= CAP; kk <<= 1) {
        for (int j = kk >> 1; j > 0; j >>= 1) {
            for (int i = threadIdx.x; i < CAP; i += 256) {
                int ixj = i ^ j;
                if (ixj > i) {
                    bool desc = ((i & kk) == 0);
                    unsigned long long a = key[i], c = key[ixj];
                    if ((a < c) == desc) { key[i] = c; key[ixj] = a; }
                }
            }
            __syncthreads();
        }
    }

    int t = threadIdx.x;
    if (t < KPAD) {
        float x1 = 1e30f, y1 = 1e30f, x2 = -1e30f, y2 = -1e30f;  // empty box
        if (t < Kn && t < n) {
            unsigned long long k = key[t];
            float    score = __uint_as_float((uint32_t)(k >> 32));
            uint32_t idx   = 0xFFFFFFFFu - (uint32_t)k;
            uint32_t p     = idx & (HWn - 1);      // index within H*W
            float xs = (float)(p & 255);
            float ys = (float)(p >> 8);
            const float* ob = off + (size_t)b * 2 * HWn;
            const float* wb = wh  + (size_t)b * 2 * HWn;
            float x  = xs + ob[p];
            float y  = ys + ob[HWn + p];
            float hw = 0.5f * wb[p];
            float hh = 0.5f * wb[HWn + p];
            if (score > TH) { x1 = x - hw; y1 = y - hh; x2 = x + hw; y2 = y + hh; }
        }
        float* bb = boxes + (size_t)b * 4 * KPAD;
        bb[t]            = x1;
        bb[KPAD + t]     = y1;
        bb[2 * KPAD + t] = x2;
        bb[3 * KPAD + t] = y2;
    }
}

// ---------------------------------------------------------------------------
// Kernel 5: rasterize union mask via WMMA. mask[h,w] = sum_k row[k,h]*col[k,w]
// per batch: [256 x 128] x [128 x 256] matmul in f16 (values 0/1, f32 accum,
// exact). One wave per 16x16 output tile; fragments generated per ISA 7.12.2
// layouts, 4 x v_wmma_f32_16x16x32_f16 per tile. EXEC uniform (no branches).
// ---------------------------------------------------------------------------
__global__ __launch_bounds__(256) void k_raster(const float* __restrict__ boxes,
                                                const float* __restrict__ image,
                                                float* __restrict__ out) {
    int b      = blockIdx.x >> 5;       // 32 blocks per batch
    int tgroup = blockIdx.x & 31;
    int wave   = threadIdx.x >> 5;      // 8 waves -> 8 tiles per block
    int lane   = threadIdx.x & 31;
    int tile   = tgroup * 8 + wave;     // 0..255
    int tm     = tile >> 4;
    int tn     = tile & 15;

    const float* bb = boxes + (size_t)b * 4 * KPAD;
    const float* X1 = bb;
    const float* Y1 = bb + KPAD;
    const float* X2 = bb + 2 * KPAD;
    const float* Y2 = bb + 3 * KPAD;

    int   hi   = lane >> 4;             // lane group (0: lanes 0-15, 1: 16-31)
    int   mn   = lane & 15;
    float hrow = (float)(tm * 16 + mn); // A fragment: M = lane&15
    float wcol = (float)(tn * 16 + mn); // B fragment: N = lane&15

    v8f acc = {};
#pragma unroll
    for (int ks = 0; ks < 4; ++ks) {
        int kbase = ks * 32;
        v16h afrag, bfrag;
#pragma unroll
        for (int e = 0; e < 16; ++e) {
            // A 16x32 f16 layout: VGPR j holds K={2j,2j+1} (j<4) / {2j+8,2j+9}
            // (j>=4) for lanes 0-15; +8 for lanes 16-31.  => K = (e<8?e:e+8)+8*hi
            int ka = kbase + ((e < 8) ? e : e + 8) + 8 * hi;
            float ay1 = Y1[ka], ay2 = Y2[ka];
            afrag[e] = (_Float16)((hrow >= ay1 && hrow <= ay2) ? 1.0f : 0.0f);
            // B 32x16 f16 layout: same K order as A.  => K = (e<8?e:e+8)+8*hi
            int kb = kbase + ((e < 8) ? e : e + 8) + 8 * hi;
            float bx1 = X1[kb], bx2 = X2[kb];
            bfrag[e] = (_Float16)((wcol >= bx1 && wcol <= bx2) ? 1.0f : 0.0f);
        }
        acc = __builtin_amdgcn_wmma_f32_16x16x32_f16(
            /*neg_a=*/false, afrag, /*neg_b=*/false, bfrag,
            /*c_mod=*/(short)0, acc, /*reuse_a=*/false, /*reuse_b=*/false);
    }

    // D layout: element r -> M = r + 8*hi, N = lane&15
    const float* img = image + (size_t)b * HWn;
    float*       o   = out   + (size_t)b * HWn;
#pragma unroll
    for (int r = 0; r < 8; ++r) {
        int h = tm * 16 + r + 8 * hi;
        int w = tn * 16 + mn;
        int p = (h << 8) | w;
        o[p] = (acc[r] > 0.0f) ? img[p] : 0.0f;
    }
}

// ---------------------------------------------------------------------------
// Host launcher
// ---------------------------------------------------------------------------
extern "C" void kernel_launch(void* const* d_in, const int* in_sizes, int n_in,
                              void* d_out, int out_size, void* d_ws, size_t ws_size,
                              hipStream_t stream) {
    const float* hm  = (const float*)d_in[0];   // [8,80,256,256]
    const float* off = (const float*)d_in[1];   // [8,2,256,256]
    const float* wh  = (const float*)d_in[2];   // [8,2,256,256]
    const float* img = (const float*)d_in[3];   // [8,256,256]
    // d_in[4] = topK (100), compile-time constant here
    float* out = (float*)d_out;                 // [8,256,256]

    // Workspace layout (u32 units). Total ~311 KB.
    uint32_t* hist = (uint32_t*)d_ws;           // Bn*NBIN
    uint32_t* thr  = hist + Bn * NBIN;          // Bn
    uint32_t* cnt  = thr + Bn;                  // Bn
    float*    cval = (float*)(cnt + Bn);        // Bn*CAP
    uint32_t* cidx = (uint32_t*)(cval + (size_t)Bn * CAP); // Bn*CAP
    float*    boxes= (float*)(cidx + (size_t)Bn * CAP);    // Bn*4*KPAD

    k_init    <<<1,            1024, 0, stream>>>(hist);
    k_nms_hist<<<Bn * Cn,      1024, 0, stream>>>(hm, hist);
    k_thresh  <<<1,            32,   0, stream>>>(hist, thr);
    k_collect <<<Bn * Cn,      1024, 0, stream>>>(hm, thr, cnt, cval, cidx);
    k_select  <<<Bn,           256,  0, stream>>>(cval, cidx, cnt, off, wh, boxes);
    k_raster  <<<Bn * 32,      256,  0, stream>>>(boxes, img, out);
}
